// ModelDecomposedExport_34213709480436
// MI455X (gfx1250) — compile-verified, matrix operand order checked by reading, not run on hardware
//
#include <hip/hip_runtime.h>
#include <hip/hip_bf16.h>

// ---------------------------------------------------------------------------
// MI455X (gfx1250) transformer forward.
// All heavy GEMMs run through v_wmma_f32_16x16x32_bf16 (wave32 WMMA).
// 128x128 block tile, 8 waves x (32x64) -> 8 WMMA per wave per K-step.
// Double-buffered LDS pipeline; uses gfx1250 async global->LDS copies
// (ASYNCcnt) when the toolchain declares the builtins, else reg staging.
// ---------------------------------------------------------------------------

#define POS_LEN 19
#define SEQ     361        // 19*19
#define SEQP    384        // SEQ padded to multiple of 32 (12 * 32)
#define CDIM    768
#define NH      12
#define HDIM    64
#define FFN     2048
#define NL      8
#define NB      32
#define BS      (NB * SEQ) // 11552 rows
#define EPS     1e-6f

typedef __attribute__((ext_vector_type(16))) __bf16 v16bf;
typedef __attribute__((ext_vector_type(8)))  __bf16 v8bf;
typedef __attribute__((ext_vector_type(8)))  float  v8f;
typedef int v4i_vs __attribute__((vector_size(16)));   // builtin's operand type

#define LDSTRIDE 40   // 32 K-halfs + 8 pad: 16B-aligned rows, no bank conflicts
#define TILEELEMS (128 * LDSTRIDE)

#if __has_builtin(__builtin_amdgcn_global_load_async_to_lds_b128) && \
    __has_builtin(__builtin_amdgcn_s_wait_asynccnt)
#define HAVE_ASYNC_LDS 1
#else
#define HAVE_ASYNC_LDS 0
#endif

#if HAVE_ASYNC_LDS
// 16-byte async global->LDS copy (GLOBAL_LOAD_ASYNC_TO_LDS_B128, ASYNCcnt).
// Param 0: AS(1) v4i* (global source), param 1: AS(3) v4i* (LDS dest).
#define GASYNC_B128(gsrc, ldst)                                              \
    __builtin_amdgcn_global_load_async_to_lds_b128(                          \
        (__attribute__((address_space(1))) v4i_vs*)(gsrc),                   \
        (__attribute__((address_space(3))) v4i_vs*)(ldst), 0, 0)
#endif

// Load a 16-half WMMA operand fragment from LDS: two 16-byte ds reads.
// CDNA5 16-bit A/B layout: lanes 0-15 hold K{0..7,16..23},
// lanes 16-31 hold K{8..15,24..31} for their M/N row.
__device__ __forceinline__ v16bf load_frag(const __bf16* p) {
    v8bf lo = *(const v8bf*)p;        // K base + 0..7
    v8bf hi = *(const v8bf*)(p + 16); // K base + 16..23
    return __builtin_shufflevector(lo, hi, 0, 1, 2, 3, 4, 5, 6, 7,
                                   8, 9, 10, 11, 12, 13, 14, 15);
}

// ---------------------------------------------------------------------------
// Batched GEMM: C[z][M,N] = A[z][M,K] * Bt[z][N,K]^T (+ optional residual)
// A, Bt bf16 row-major; C f32. K must be a multiple of 32.
// Out-of-range tile rows are CLAMPED (not zeroed): they only feed outputs
// that are never stored, so garbage there is harmless and the hot loop has
// no EXEC divergence.
// C base address for batch z: (z / zdiv) * sCb + (z % zdiv) * sCh.
// ---------------------------------------------------------------------------
__global__ __launch_bounds__(256) void gemm_bf16_wmma(
    const __bf16* __restrict__ A, const __bf16* __restrict__ Bt,
    float* __restrict__ C, const float* __restrict__ Res,
    int M, int N, int K, int lda, int ldb, int ldc,
    long long sAz, long long sBz, int zdiv, long long sCb, long long sCh)
{
    __shared__ __align__(16) __bf16 sA[2 * TILEELEMS]; // double-buffered
    __shared__ __align__(16) __bf16 sB[2 * TILEELEMS];

    const int tid  = threadIdx.x;
    const int lane = tid & 31;
    const int wave = tid >> 5;     // 8 waves
    const int wr   = wave & 3;     // wave row group 0..3  (32 rows each)
    const int wc   = wave >> 2;    // wave col group 0..1  (64 cols each)

    const int n0 = blockIdx.x * 128;
    const int m0 = blockIdx.y * 128;
    const int z  = blockIdx.z;

    const __bf16* Az = A  + (size_t)z * (size_t)sAz;
    const __bf16* Bz = Bt + (size_t)z * (size_t)sBz;
    const size_t cbase = (size_t)(z / zdiv) * (size_t)sCb +
                         (size_t)(z % zdiv) * (size_t)sCh;

    // Staging: 128x32 tile = 512 16B chunks; 256 threads x 2 chunks.
    const int c0row = tid >> 2;
    const int c1row = (tid + 256) >> 2;
    const int cseg  = (tid & 3) * 8;          // same for both chunks
    // Branchless clamped source rows (no EXEC divergence).
    const int aRow0 = (m0 + c0row < M) ? (m0 + c0row) : (M - 1);
    const int aRow1 = (m0 + c1row < M) ? (m0 + c1row) : (M - 1);
    const int bRow0 = (n0 + c0row < N) ? (n0 + c0row) : (N - 1);
    const int bRow1 = (n0 + c1row < N) ? (n0 + c1row) : (N - 1);

    const __bf16* aSrc0 = Az + (size_t)aRow0 * lda + cseg;
    const __bf16* aSrc1 = Az + (size_t)aRow1 * lda + cseg;
    const __bf16* bSrc0 = Bz + (size_t)bRow0 * ldb + cseg;
    const __bf16* bSrc1 = Bz + (size_t)bRow1 * ldb + cseg;
    const int dst0 = c0row * LDSTRIDE + cseg;
    const int dst1 = c1row * LDSTRIDE + cseg;

    const int r16 = lane & 15;
    const int kh  = (lane >> 4) * 8;   // 0 or 8

    v8f acc[2][4];
#pragma unroll
    for (int i = 0; i < 2; ++i)
#pragma unroll
        for (int j = 0; j < 4; ++j) acc[i][j] = (v8f){};

#if HAVE_ASYNC_LDS
    // ---- async global->LDS pipeline (no VGPR roundtrip) -------------------
    GASYNC_B128(aSrc0, &sA[dst0]);
    GASYNC_B128(aSrc1, &sA[dst1]);
    GASYNC_B128(bSrc0, &sB[dst0]);
    GASYNC_B128(bSrc1, &sB[dst1]);
    __builtin_amdgcn_s_wait_asynccnt(0);
    __syncthreads();
#else
    // ---- register-staged prologue -----------------------------------------
    {
        const uint4 ra0 = *(const uint4*)aSrc0;
        const uint4 ra1 = *(const uint4*)aSrc1;
        const uint4 rb0 = *(const uint4*)bSrc0;
        const uint4 rb1 = *(const uint4*)bSrc1;
        *(uint4*)&sA[dst0] = ra0;
        *(uint4*)&sA[dst1] = ra1;
        *(uint4*)&sB[dst0] = rb0;
        *(uint4*)&sB[dst1] = rb1;
    }
    __syncthreads();
#endif

    for (int k0 = 0; k0 < K; k0 += 32) {
        const int  cur  = (k0 >> 5) & 1;
        const int  nxt  = cur ^ 1;
        const bool more = (k0 + 32) < K;
        const int  kn   = k0 + 32;

#if HAVE_ASYNC_LDS
        if (more) { // issue next tile's async copies; waited after compute
            GASYNC_B128(aSrc0 + kn, &sA[nxt * TILEELEMS + dst0]);
            GASYNC_B128(aSrc1 + kn, &sA[nxt * TILEELEMS + dst1]);
            GASYNC_B128(bSrc0 + kn, &sB[nxt * TILEELEMS + dst0]);
            GASYNC_B128(bSrc1 + kn, &sB[nxt * TILEELEMS + dst1]);
        }
#else
        uint4 ra0, ra1, rb0, rb1;
        if (more) { // 4 independent outstanding loads
            ra0 = *(const uint4*)(aSrc0 + kn);
            ra1 = *(const uint4*)(aSrc1 + kn);
            rb0 = *(const uint4*)(bSrc0 + kn);
            rb1 = *(const uint4*)(bSrc1 + kn);
        }
#endif

        // ---- compute from current buffer ----------------------------------
        const __bf16* bufA = sA + cur * TILEELEMS;
        const __bf16* bufB = sB + cur * TILEELEMS;
        v16bf af[2], bfr[4];
#pragma unroll
        for (int i = 0; i < 2; ++i)
            af[i] = load_frag(&bufA[(wr * 32 + i * 16 + r16) * LDSTRIDE + kh]);
#pragma unroll
        for (int j = 0; j < 4; ++j)
            bfr[j] = load_frag(&bufB[(wc * 64 + j * 16 + r16) * LDSTRIDE + kh]);

#pragma unroll
        for (int i = 0; i < 2; ++i)
#pragma unroll
            for (int j = 0; j < 4; ++j)
                acc[i][j] = __builtin_amdgcn_wmma_f32_16x16x32_bf16(
                    false, af[i], false, bfr[j], (short)0, acc[i][j], false, false);

        // ---- land next tile into the other buffer --------------------------
#if HAVE_ASYNC_LDS
        if (more) __builtin_amdgcn_s_wait_asynccnt(0);
#else
        if (more) {
            *(uint4*)&sA[nxt * TILEELEMS + dst0] = ra0;
            *(uint4*)&sA[nxt * TILEELEMS + dst1] = ra1;
            *(uint4*)&sB[nxt * TILEELEMS + dst0] = rb0;
            *(uint4*)&sB[nxt * TILEELEMS + dst1] = rb1;
        }
#endif
        __syncthreads();
    }

    // C/D layout: VGPR v -> M = v + (lane>=16 ? 8 : 0); N = lane & 15
    const int mbase = m0 + wr * 32 + (lane >> 4) * 8;
    const int nbase = n0 + wc * 64 + (lane & 15);
#pragma unroll
    for (int i = 0; i < 2; ++i) {
#pragma unroll
        for (int j = 0; j < 4; ++j) {
            const int n = nbase + j * 16;
            if (n >= N) continue;
#pragma unroll
            for (int v = 0; v < 8; ++v) {
                const int m = mbase + i * 16 + v;
                if (m < M) {
                    const size_t o = cbase + (size_t)m * ldc + n;
                    float val = acc[i][j][v];
                    if (Res) val += Res[o];
                    C[o] = val;
                }
            }
        }
    }
}

// ---------------------------------------------------------------------------
// Weight convert + transpose: fp32 src[K,N] -> bf16 dst[N,K]
// ---------------------------------------------------------------------------
__global__ __launch_bounds__(256) void wtcvt_kernel(
    const float* __restrict__ src, __bf16* __restrict__ dst, int K, int N)
{
    const long long i = (long long)blockIdx.x * 256 + threadIdx.x;
    if (i >= (long long)K * N) return;
    const int k = (int)(i / N);
    const int n = (int)(i % N);
    dst[(size_t)n * K + k] = (__bf16)src[i];
}

// flat fp32 -> bf16
__global__ __launch_bounds__(256) void cvt_bf16_kernel(
    const float* __restrict__ src, __bf16* __restrict__ dst, long long n)
{
    const long long i = (long long)blockIdx.x * 256 + threadIdx.x;
    if (i < n) dst[i] = (__bf16)src[i];
}

// ---------------------------------------------------------------------------
// x_g[b,c] = input_global[b,:] @ Wg[:,c]
// ---------------------------------------------------------------------------
__global__ __launch_bounds__(256) void xg_kernel(
    const float* __restrict__ ig, const float* __restrict__ Wg,
    float* __restrict__ xg)
{
    const int idx = blockIdx.x * 256 + threadIdx.x;
    if (idx >= NB * CDIM) return;
    const int b = idx / CDIM, c = idx % CDIM;
    float s = 0.f;
#pragma unroll
    for (int g = 0; g < POS_LEN; ++g)
        s += ig[b * POS_LEN + g] * Wg[g * CDIM + c];
    xg[idx] = s;
}

// 3x3 SAME conv (22 -> 768 ch) + broadcast global embed; out [B, SEQ, C]
__global__ __launch_bounds__(256) void conv_embed_kernel(
    const float* __restrict__ inp, const float* __restrict__ w,
    const float* __restrict__ xg, float* __restrict__ x)
{
    const long long idx = (long long)blockIdx.x * 256 + threadIdx.x;
    if (idx >= (long long)NB * SEQ * CDIM) return;
    const int c = (int)(idx % CDIM);
    const int s = (int)((idx / CDIM) % SEQ);
    const int b = (int)(idx / ((long long)CDIM * SEQ));
    const int row = s / POS_LEN, col = s % POS_LEN;

    float sum = xg[b * CDIM + c];
    const float* wc = w + (size_t)c * 22 * 9;
    const float* ib = inp + (size_t)b * 22 * SEQ;
    for (int ci = 0; ci < 22; ++ci) {
        const float* ip = ib + ci * SEQ;
        const float* wp = wc + ci * 9;
#pragma unroll
        for (int kh = 0; kh < 3; ++kh) {
            const int ir = row + kh - 1;
            if (ir < 0 || ir >= POS_LEN) continue;
#pragma unroll
            for (int kw = 0; kw < 3; ++kw) {
                const int ic = col + kw - 1;
                if (ic < 0 || ic >= POS_LEN) continue;
                sum += ip[ir * POS_LEN + ic] * wp[kh * 3 + kw];
            }
        }
    }
    x[((size_t)b * SEQ + s) * CDIM + c] = sum;
}

// ---------------------------------------------------------------------------
// RMSNorm over 768 cols; optional bf16 and/or f32 outputs. One block per row.
// ---------------------------------------------------------------------------
__global__ __launch_bounds__(256) void rmsnorm_kernel(
    const float* __restrict__ x, const float* __restrict__ gamma,
    __bf16* __restrict__ outB, float* __restrict__ outF)
{
    __shared__ float red[256];
    const int r = blockIdx.x;
    const int tid = threadIdx.x;
    const float* row = x + (size_t)r * CDIM;
    const float v0 = row[tid], v1 = row[tid + 256], v2 = row[tid + 512];
    red[tid] = v0 * v0 + v1 * v1 + v2 * v2;
    __syncthreads();
    for (int s2 = 128; s2 > 0; s2 >>= 1) {
        if (tid < s2) red[tid] += red[tid + s2];
        __syncthreads();
    }
    const float rinv = __frsqrt_rn(red[0] * (1.0f / CDIM) + EPS);
    const float o0 = v0 * rinv * gamma[tid];
    const float o1 = v1 * rinv * gamma[tid + 256];
    const float o2 = v2 * rinv * gamma[tid + 512];
    if (outB) {
        __bf16* ob = outB + (size_t)r * CDIM;
        ob[tid] = (__bf16)o0; ob[tid + 256] = (__bf16)o1; ob[tid + 512] = (__bf16)o2;
    }
    if (outF) {
        float* of = outF + (size_t)r * CDIM;
        of[tid] = o0; of[tid + 256] = o1; of[tid + 512] = o2;
    }
}

// ---------------------------------------------------------------------------
// RoPE + repack: qkv f32 [B,S,3,H,HD] ->
//   qp,kp bf16 [B*H, S, HD] (rope applied), vt bf16 [B*H, HD, SEQP] (zero pad)
// ---------------------------------------------------------------------------
__global__ __launch_bounds__(256) void rope_pack_kernel(
    const float* __restrict__ qkv, __bf16* __restrict__ qp,
    __bf16* __restrict__ kp, __bf16* __restrict__ vt)
{
    const long long idx = (long long)blockIdx.x * 256 + threadIdx.x;
    const long long total = (long long)NB * NH * SEQP * HDIM;
    if (idx >= total) return;
    const int d = (int)(idx & 63);
    const int s = (int)((idx >> 6) % SEQP);
    const int h = (int)((idx / (64LL * SEQP)) % NH);
    const int b = (int)(idx / (64LL * SEQP * NH));
    const int bh = b * NH + h;

    if (s >= SEQ) { // zero-pad v^T K dimension
        vt[((size_t)bh * HDIM + d) * SEQP + s] = (__bf16)0.f;
        return;
    }
    const size_t base = ((size_t)(b * SEQ + s)) * (3 * CDIM) + h * HDIM;
    const float qv = qkv[base + d];
    const float kv = qkv[base + CDIM + d];
    const float vv = qkv[base + 2 * CDIM + d];

    const int p = (d < 32) ? d + 32 : d - 32;
    const float sgn = (d < 32) ? -1.f : 1.f;
    const float qpar = qkv[base + p];
    const float kpar = qkv[base + CDIM + p];

    const int rpos = s / POS_LEN, cpos = s % POS_LEN;
    const int j  = d & 31;
    const int fi = j & 15;
    const float invf = __powf(10000.f, -(float)fi * (1.f / 16.f));
    const float ang = ((j < 16) ? (float)rpos : (float)cpos) * invf;
    const float cs = __cosf(ang), sn = __sinf(ang);

    const size_t so = ((size_t)bh * SEQ + s) * HDIM + d;
    qp[so] = (__bf16)(qv * cs + sgn * qpar * sn);
    kp[so] = (__bf16)(kv * cs + sgn * kpar * sn);
    vt[((size_t)bh * HDIM + d) * SEQP + s] = (__bf16)vv;
}

// ---------------------------------------------------------------------------
// Softmax over 361 cols of scores[row, 0:384] (scale 1/8 folded in);
// writes bf16 attn with zero pad in cols 361..383. One block per row.
// ---------------------------------------------------------------------------
__global__ __launch_bounds__(256) void softmax_kernel(
    const float* __restrict__ scores, __bf16* __restrict__ attn)
{
    __shared__ float red[256];
    const size_t row = blockIdx.x;
    const float* src = scores + row * SEQP;
    __bf16* dst = attn + row * SEQP;
    const int tid = threadIdx.x;

    const float a = (tid < SEQ) ? src[tid] * 0.125f : -3.4e38f;
    const float b = (tid + 256 < SEQ) ? src[tid + 256] * 0.125f : -3.4e38f;
    red[tid] = fmaxf(a, b);
    __syncthreads();
    for (int s2 = 128; s2 > 0; s2 >>= 1) {
        if (tid < s2) red[tid] = fmaxf(red[tid], red[tid + s2]);
        __syncthreads();
    }
    const float mx = red[0];
    __syncthreads();

    const float ea = (tid < SEQ) ? __expf(a - mx) : 0.f;
    const float eb = (tid + 256 < SEQ) ? __expf(b - mx) : 0.f;
    red[tid] = ea + eb;
    __syncthreads();
    for (int s2 = 128; s2 > 0; s2 >>= 1) {
        if (tid < s2) red[tid] += red[tid + s2];
        __syncthreads();
    }
    const float inv = 1.f / red[0];

    if (tid < SEQ) dst[tid] = (__bf16)(ea * inv);
    const int j = tid + 256;
    if (j < SEQP) dst[j] = (j < SEQ) ? (__bf16)(eb * inv) : (__bf16)0.f;
}

// SwiGLU: u[R, 4096] -> g[R, 2048] bf16, g = silu(u[:, :2048]) * u[:, 2048:]
__global__ __launch_bounds__(256) void swiglu_kernel(
    const float* __restrict__ u, __bf16* __restrict__ g)
{
    const long long i = (long long)blockIdx.x * 256 + threadIdx.x;
    if (i >= (long long)BS * FFN) return;
    const long long r = i / FFN;
    const int j = (int)(i % FFN);
    const float a = u[r * (2 * FFN) + j];
    const float b = u[r * (2 * FFN) + FFN + j];
    const float sg = a / (1.f + __expf(-a));
    g[i] = (__bf16)(sg * b);
}

// ---------------------------------------------------------------------------
// Host launcher
// ---------------------------------------------------------------------------
extern "C" void kernel_launch(void* const* d_in, const int* in_sizes, int n_in,
                              void* d_out, int out_size, void* d_ws, size_t ws_size,
                              hipStream_t stream)
{
    const float* input_spatial = (const float*)d_in[0];
    const float* input_global  = (const float*)d_in[1];
    const float* conv_w        = (const float*)d_in[2];
    const float* Wg            = (const float*)d_in[3];
    const float* qkv_gamma     = (const float*)d_in[4];
    const float* Wqkv          = (const float*)d_in[5];
    const float* Wproj         = (const float*)d_in[6];
    const float* mlp_gamma     = (const float*)d_in[7];
    const float* Wfc1          = (const float*)d_in[8];
    const float* Wfc2          = (const float*)d_in[9];
    const float* final_gamma   = (const float*)d_in[10];
    (void)in_sizes; (void)n_in; (void)out_size; (void)ws_size;

    char* ws = (char*)d_ws;
    size_t off = 0;
    auto alloc = [&](size_t bytes) -> void* {
        off = (off + 255) & ~(size_t)255;
        void* p = ws + off;
        off += bytes;
        return p;
    };

    const int BH = NB * NH; // 384 attention batches

    __bf16* wqkvT  = (__bf16*)alloc((size_t)NL * 3 * CDIM * CDIM * sizeof(__bf16));
    __bf16* wprojT = (__bf16*)alloc((size_t)NL * CDIM * CDIM * sizeof(__bf16));
    __bf16* wfc1T  = (__bf16*)alloc((size_t)NL * 2 * FFN * CDIM * sizeof(__bf16));
    __bf16* wfc2T  = (__bf16*)alloc((size_t)NL * CDIM * FFN * sizeof(__bf16));
    float*  x      = (float*)alloc((size_t)BS * CDIM * sizeof(float));
    float*  xg     = (float*)alloc((size_t)NB * CDIM * sizeof(float));
    __bf16* h      = (__bf16*)alloc((size_t)BS * CDIM * sizeof(__bf16));
    float*  qkv    = (float*)alloc((size_t)BS * 3 * CDIM * sizeof(float));
    __bf16* qp     = (__bf16*)alloc((size_t)BH * SEQ * HDIM * sizeof(__bf16));
    __bf16* kp     = (__bf16*)alloc((size_t)BH * SEQ * HDIM * sizeof(__bf16));
    __bf16* vt     = (__bf16*)alloc((size_t)BH * HDIM * SEQP * sizeof(__bf16));
    float*  scores = (float*)alloc((size_t)BH * SEQ * SEQP * sizeof(float));
    __bf16* attn   = (__bf16*)alloc((size_t)BH * SEQ * SEQP * sizeof(__bf16));
    float*  o      = (float*)alloc((size_t)BS * CDIM * sizeof(float));
    __bf16* obf    = (__bf16*)alloc((size_t)BS * CDIM * sizeof(__bf16));
    float*  u      = (float*)alloc((size_t)BS * 2 * FFN * sizeof(float));
    __bf16* g      = (__bf16*)alloc((size_t)BS * FFN * sizeof(__bf16));

    auto grid1 = [](long long n) { return dim3((unsigned)((n + 255) / 256)); };

    auto gemm = [&](const __bf16* A, const __bf16* Bt, float* Cp, const float* Res,
                    int M, int N, int K, int lda, int ldb, int ldc,
                    long long sAz, long long sBz, int zdiv,
                    long long sCb, long long sCh, int Z) {
        dim3 gdim((N + 127) / 128, (M + 127) / 128, Z);
        gemm_bf16_wmma<<<gdim, 256, 0, stream>>>(A, Bt, Cp, Res, M, N, K,
                                                 lda, ldb, ldc, sAz, sBz,
                                                 zdiv, sCb, sCh);
    };

    // --- one-time weight conversion (fp32 [K,N] -> bf16 [N,K]) -------------
    for (int i = 0; i < NL; ++i) {
        wtcvt_kernel<<<grid1((long long)CDIM * 3 * CDIM), 256, 0, stream>>>(
            Wqkv + (size_t)i * CDIM * 3 * CDIM, wqkvT + (size_t)i * 3 * CDIM * CDIM,
            CDIM, 3 * CDIM);
        wtcvt_kernel<<<grid1((long long)CDIM * CDIM), 256, 0, stream>>>(
            Wproj + (size_t)i * CDIM * CDIM, wprojT + (size_t)i * CDIM * CDIM,
            CDIM, CDIM);
        wtcvt_kernel<<<grid1((long long)CDIM * 2 * FFN), 256, 0, stream>>>(
            Wfc1 + (size_t)i * CDIM * 2 * FFN, wfc1T + (size_t)i * 2 * FFN * CDIM,
            CDIM, 2 * FFN);
        wtcvt_kernel<<<grid1((long long)FFN * CDIM), 256, 0, stream>>>(
            Wfc2 + (size_t)i * FFN * CDIM, wfc2T + (size_t)i * CDIM * FFN,
            FFN, CDIM);
    }

    // --- stem: conv + global embed -> x [B, SEQ, C] ------------------------
    xg_kernel<<<grid1((long long)NB * CDIM), 256, 0, stream>>>(input_global, Wg, xg);
    conv_embed_kernel<<<grid1((long long)NB * SEQ * CDIM), 256, 0, stream>>>(
        input_spatial, conv_w, xg, x);

    // --- transformer layers ------------------------------------------------
    for (int i = 0; i < NL; ++i) {
        // h = rmsnorm(x) (bf16)
        rmsnorm_kernel<<<BS, 256, 0, stream>>>(x, qkv_gamma + (size_t)i * CDIM, h, nullptr);

        // qkv = h @ Wqkv  (f32 [BS, 2304])
        gemm(h, wqkvT + (size_t)i * 3 * CDIM * CDIM, qkv, nullptr,
             BS, 3 * CDIM, CDIM, CDIM, CDIM, 3 * CDIM, 0, 0, 1, 0, 0, 1);

        // rope + pack q/k (bf16) and v^T (bf16, zero-padded to SEQP)
        rope_pack_kernel<<<grid1((long long)NB * NH * SEQP * HDIM), 256, 0, stream>>>(
            qkv, qp, kp, vt);

        // scores[bh] = q @ k^T  (f32, ld = SEQP)
        gemm(qp, kp, scores, nullptr, SEQ, SEQ, HDIM, HDIM, HDIM, SEQP,
             (long long)SEQ * HDIM, (long long)SEQ * HDIM, 1,
             (long long)SEQ * SEQP, 0, BH);

        // softmax (scale folded) -> attn bf16 with zero pad
        softmax_kernel<<<BH * SEQ, 256, 0, stream>>>(scores, attn);

        // o[b,s,h,:] = attn[bh] @ v[bh]  (K = SEQP, padded with zeros)
        gemm(attn, vt, o, nullptr, SEQ, HDIM, SEQP, SEQP, SEQP, CDIM,
             (long long)SEQ * SEQP, (long long)HDIM * SEQP, NH,
             (long long)SEQ * CDIM, (long long)HDIM, BH);

        // x = x + o @ Wproj
        cvt_bf16_kernel<<<grid1((long long)BS * CDIM), 256, 0, stream>>>(
            o, obf, (long long)BS * CDIM);
        gemm(obf, wprojT + (size_t)i * CDIM * CDIM, x, x,
             BS, CDIM, CDIM, CDIM, CDIM, CDIM, 0, 0, 1, 0, 0, 1);

        // MLP: h = rmsnorm(x); u = h @ Wfc1; g = swiglu(u); x = x + g @ Wfc2
        rmsnorm_kernel<<<BS, 256, 0, stream>>>(x, mlp_gamma + (size_t)i * CDIM, h, nullptr);
        gemm(h, wfc1T + (size_t)i * 2 * FFN * CDIM, u, nullptr,
             BS, 2 * FFN, CDIM, CDIM, CDIM, 2 * FFN, 0, 0, 1, 0, 0, 1);
        swiglu_kernel<<<grid1((long long)BS * FFN), 256, 0, stream>>>(u, g);
        gemm(g, wfc2T + (size_t)i * CDIM * FFN, x, x,
             BS, CDIM, FFN, FFN, FFN, CDIM, 0, 0, 1, 0, 0, 1);
    }

    // --- final RMSNorm -> d_out (f32) --------------------------------------
    rmsnorm_kernel<<<BS, 256, 0, stream>>>(x, final_gamma, nullptr, (float*)d_out);
}
